// TensorF_89240830476846
// MI455X (gfx1250) — matrix operand order repaired and checked by reading, not compiled, not run on hardware
//
#include <hip/hip_runtime.h>

// CDNA5 / gfx1250 TensoRF-style fused kernel:
//   L2-resident bilinear gathers -> f16 features in LDS ->
//   v_wmma_f32_16x16x32_f16 (K=32=channels) for feat@W1 ->
//   relu + W2-dot on C-fragments + cross-lane reduce.
typedef __attribute__((ext_vector_type(16))) _Float16 v16h;
typedef __attribute__((ext_vector_type(8)))  _Float16 v8h;
typedef __attribute__((ext_vector_type(8)))  float    v8f;

#define CL 32
#define NL 512
#define HIDDEN 128
#define LDS_STRIDE 40   // halves per point = 80 bytes; multiple of 16B for b128 chunks
#define CHUNK 4         // channels per gather chunk; 4*1MB plane span < 8.3MB IOFFSET range

__global__ __launch_bounds__(256) void tensorf_fused_kernel(
    const float* __restrict__ coord,
    const float* __restrict__ line_x,
    const float* __restrict__ line_y,
    const float* __restrict__ line_z,
    const float* __restrict__ plane_xy,
    const float* __restrict__ plane_yz,
    const float* __restrict__ plane_xz,
    const float* __restrict__ W1,
    const float* __restrict__ b1,
    const float* __restrict__ W2,
    const float* __restrict__ b2,
    float* __restrict__ out,
    int P, int n_groups)
{
    __shared__ __align__(16) _Float16 smem[8][32][LDS_STRIDE];  // 20 KB feature tiles
    __shared__ __align__(16) _Float16 w1s[HIDDEN * CL];         // 8 KB: w1s[n*32+k] = W1[k][n]

    const int lane = threadIdx.x & 31;
    const int wave = threadIdx.x >> 5;
    const int col  = lane & 15;      // N column within a 16-wide tile
    const int half = lane >> 4;      // B: K-half; C: M 0..7 vs 8..15
    const int kb   = half * 8;

    // ---- stage W1 transposed to f16 in LDS (once per block, shared by 8 waves) ----
    for (int i = threadIdx.x; i < HIDDEN * CL; i += 256) {
        const int n = i >> 5, k = i & 31;
        w1s[i] = (_Float16)W1[k * HIDDEN + n];
    }
    __syncthreads();

    // ---- small per-lane constants: b1 / W2 columns this lane owns, b2 ----
    float biasv[8], w2v[8];
#pragma unroll
    for (int n0 = 0; n0 < 8; ++n0) {
        const int n = n0 * 16 + col;
        biasv[n0] = b1[n];
        w2v[n0]   = W2[n];
    }
    const float b2v = b2[0];

    const int wave_id    = blockIdx.x * 8 + wave;
    const int wave_count = gridDim.x * 8;

    for (int g = wave_id; g < n_groups; g += wave_count) {
        // ---------- phase A: per-point feature gather (lane = point) ----------
        const int p = min(g * 32 + lane, P - 1);
        const float x = coord[p * 3 + 0];
        const float y = coord[p * 3 + 1];
        const float z = coord[p * 3 + 2];

        const float fx = (x + 1.0f) * (0.5f * (NL - 1));
        const float fy = (y + 1.0f) * (0.5f * (NL - 1));
        const float fz = (z + 1.0f) * (0.5f * (NL - 1));
        // clamp cell to [0, NL-2]: at the exact upper edge w1==1 selects element NL-1,
        // numerically identical to the reference's zero-padded corner (weight 0 there).
        const int xi = min(max((int)floorf(fx), 0), NL - 2);
        const int yi = min(max((int)floorf(fy), 0), NL - 2);
        const int zi = min(max((int)floorf(fz), 0), NL - 2);
        const float wx1 = fx - (float)xi, wx0 = 1.0f - wx1;
        const float wy1 = fy - (float)yi, wy0 = 1.0f - wy1;
        const float wz1 = fz - (float)zi, wz0 = 1.0f - wz1;

        // bilinear corner weights (H-weight x W-weight per the reference's gx/gy mapping)
        const float cxy00 = wy0*wx0, cxy01 = wy0*wx1, cxy10 = wy1*wx0, cxy11 = wy1*wx1;
        const float cyz00 = wz0*wy0, cyz01 = wz0*wy1, cyz10 = wz1*wy0, cyz11 = wz1*wy1;
        const float cxz00 = wz0*wx0, cxz01 = wz0*wx1, cxz10 = wz1*wx0, cxz11 = wz1*wx1;

        const float* lxp  = line_x  + xi;
        const float* lyp  = line_y  + yi;
        const float* lzp  = line_z  + zi;
        const float* pxyp = plane_xy + yi * NL + xi;   // [c, y, x]
        const float* pyzp = plane_yz + zi * NL + yi;   // [c, z, y]
        const float* pxzp = plane_xz + zi * NL + xi;   // [c, z, x]

        _Float16* fdst = &smem[wave][lane][0];

        for (int c0 = 0; c0 < CL; c0 += CHUNK) {
            // one 64-bit base add per pointer per chunk; all loads below use
            // compile-time immediate offsets -> large load clauses, batched waits
            const float* bxy = pxyp + (size_t)c0 * (NL * NL);
            const float* byz = pyzp + (size_t)c0 * (NL * NL);
            const float* bxz = pxzp + (size_t)c0 * (NL * NL);
            const float* blx = lxp + c0 * NL;
            const float* bly = lyp + c0 * NL;
            const float* blz = lzp + c0 * NL;

            float vl[CHUNK][6];
            float vp[CHUNK][12];
#pragma unroll
            for (int u = 0; u < CHUNK; ++u) {          // issue all 18*CHUNK loads first
                const int lo = u * NL;
                const int po = u * (NL * NL);
                vl[u][0] = blx[lo];        vl[u][1] = blx[lo + 1];
                vl[u][2] = bly[lo];        vl[u][3] = bly[lo + 1];
                vl[u][4] = blz[lo];        vl[u][5] = blz[lo + 1];
                vp[u][0] = bxy[po];        vp[u][1] = bxy[po + 1];
                vp[u][2] = bxy[po + NL];   vp[u][3] = bxy[po + NL + 1];
                vp[u][4] = byz[po];        vp[u][5] = byz[po + 1];
                vp[u][6] = byz[po + NL];   vp[u][7] = byz[po + NL + 1];
                vp[u][8] = bxz[po];        vp[u][9] = bxz[po + 1];
                vp[u][10] = bxz[po + NL];  vp[u][11] = bxz[po + NL + 1];
            }
#pragma unroll
            for (int u = 0; u < CHUNK; ++u) {          // then consume
                const float lx = vl[u][0] * wx0 + vl[u][1] * wx1;
                const float ly = vl[u][2] * wy0 + vl[u][3] * wy1;
                const float lz = vl[u][4] * wz0 + vl[u][5] * wz1;
                const float vxy = vp[u][0] * cxy00 + vp[u][1] * cxy01
                                + vp[u][2] * cxy10 + vp[u][3] * cxy11;
                const float vyz = vp[u][4] * cyz00 + vp[u][5] * cyz01
                                + vp[u][6] * cyz10 + vp[u][7] * cyz11;
                const float vxz = vp[u][8] * cxz00 + vp[u][9] * cxz01
                                + vp[u][10] * cxz10 + vp[u][11] * cxz11;
                fdst[c0 + u] = (_Float16)(lx * vyz + ly * vxz + lz * vxy);
            }
        }
        // Wave-private feature region + DS ops from one wave complete in order:
        // no block barrier needed before re-reading (w1s is read-only after init).

        // ---------- phase B: two 16-point WMMA tiles ----------
        const _Float16* w1base = &w1s[col * CL + kb];  // this lane's B column/K-half
#pragma unroll
        for (int t = 0; t < 2; ++t) {
            // A 16x32 f16 layout: lane<16 -> point=col, K {0..7,16..23};
            //                     lane>=16 -> point=col, K {8..15,24..31}
            const _Float16* src = &smem[wave][t * 16 + col][0];
            const int chb = half * 8;
            const v8h a0 = *(const v8h*)(src + chb);        // ds_load_b128
            const v8h a1 = *(const v8h*)(src + chb + 16);   // ds_load_b128
            v16h a;
#pragma unroll
            for (int e = 0; e < 8; ++e) { a[e] = a0[e]; a[8 + e] = a1[e]; }

            const v8f czero = {};
            v8f acc = {};
#pragma unroll
            for (int n0 = 0; n0 < 8; ++n0) {
                // B 32x16 f16: lane<16 holds K {0..7,16..23} of col n0*16+col, lane>=16 K {8..15,24..31}
                const v8h b0 = *(const v8h*)(w1base + n0 * (16 * CL));
                const v8h b1f = *(const v8h*)(w1base + n0 * (16 * CL) + 16);
                v16h b;
#pragma unroll
                for (int e = 0; e < 8; ++e) { b[e] = b0[e]; b[8 + e] = b1f[e]; }

                // D(16x16 f32) = A(16x32 f16) x W1-tile(32x16 f16)
                v8f h = __builtin_amdgcn_wmma_f32_16x16x32_f16(
                    false, a, false, b, (short)0, czero, false, false);
                const float bb = biasv[n0], ww = w2v[n0];
#pragma unroll
                for (int r = 0; r < 8; ++r) {
                    const float hv = fmaxf(h[r] + bb, 0.0f);
                    acc[r] = fmaf(hv, ww, acc[r]);   // layer-2 dot, per-column partial
                }
            }
            // reduce the 16 columns inside each 16-lane half (xor 1,2,4,8 never
            // crosses the half boundary); lane 0 -> points t*16+0..7, lane 16 -> +8..15
#pragma unroll
            for (int r = 0; r < 8; ++r) {
                float v = acc[r];
                v += __shfl_xor(v, 1, 32);
                v += __shfl_xor(v, 2, 32);
                v += __shfl_xor(v, 4, 32);
                v += __shfl_xor(v, 8, 32);
                acc[r] = v;
            }
            if (col == 0) {
                const int pbase = g * 32 + t * 16 + half * 8;
#pragma unroll
                for (int r = 0; r < 8; ++r) {
                    const int po = pbase + r;
                    if (po < P) out[po] = acc[r] + b2v;
                }
            }
        }
    }
}

extern "C" void kernel_launch(void* const* d_in, const int* in_sizes, int n_in,
                              void* d_out, int out_size, void* d_ws, size_t ws_size,
                              hipStream_t stream) {
    const float* coord    = (const float*)d_in[0];
    const float* line_x   = (const float*)d_in[1];
    const float* line_y   = (const float*)d_in[2];
    const float* line_z   = (const float*)d_in[3];
    const float* plane_xy = (const float*)d_in[4];
    const float* plane_yz = (const float*)d_in[5];
    const float* plane_xz = (const float*)d_in[6];
    const float* W1       = (const float*)d_in[7];
    const float* b1       = (const float*)d_in[8];
    const float* W2       = (const float*)d_in[9];
    const float* b2       = (const float*)d_in[10];
    float* out = (float*)d_out;

    const int P        = in_sizes[0] / 3;
    const int n_groups = (P + 31) / 32;          // 32 points per wave per pass
    int blocks = (n_groups + 7) / 8;             // 8 waves per block
    if (blocks > 2048) blocks = 2048;            // grid-stride; amortizes LDS/reg init
    if (blocks < 1) blocks = 1;

    tensorf_fused_kernel<<<blocks, 256, 0, stream>>>(
        coord, line_x, line_y, line_z, plane_xy, plane_yz, plane_xz,
        W1, b1, W2, b2, out, P, n_groups);
}
